// BoxPromptFilter_65360812311052
// MI455X (gfx1250) — matrix operand order repaired and checked by reading, not compiled, not run on hardware
//
#include <hip/hip_runtime.h>
#include <hip/hip_bf16.h>

typedef __attribute__((ext_vector_type(16))) _Float16 v16h;
typedef __attribute__((ext_vector_type(8)))  float    v8f;

#define MBOX     1024
#define NTHREADS 256
#define THRESH   0.8f

__global__ __launch_bounds__(NTHREADS)
void BoxPromptFilter_kernel(const float* __restrict__ boxes,
                            const int*   __restrict__ nvalid,
                            float*       __restrict__ out_boxes,
                            int*         __restrict__ out_cnt)
{
    // Original-order data
    __shared__ __attribute__((aligned(16))) float ox1[MBOX], oy1[MBOX], ox2[MBOX], oy2[MBOX], osc[MBOX], oar[MBOX];
    // Sorted-order data
    __shared__ __attribute__((aligned(16))) float sx1[MBOX], sy1[MBOX], sx2[MBOX], sy2[MBOX], sa[MBOX];
    __shared__ __attribute__((aligned(16))) float skey[MBOX];
    __shared__ __attribute__((aligned(16))) int   sidx[MBOX];
    __shared__ __attribute__((aligned(16))) int   svv[MBOX];
    __shared__ int scan[NTHREADS];

    // Aliased reuse (phases are barrier-separated):
    int* keepS = (int*)oar;    // keep flag at SORTED position (oar dead after gather)
    int* keepO = (int*)skey;   // keep flag at ORIGINAL position (skey dead after sort)

    const int g   = blockIdx.x;
    const int tid = threadIdx.x;
    const int n   = nvalid[g];
    const float* gb = boxes + (size_t)g * MBOX * 5;

    // ---- Phase A: load boxes, areas, sort keys ----
    for (int i = tid; i < MBOX; i += NTHREADS) {
        __builtin_prefetch(gb + (i + NTHREADS) * 5, 0, 0);   // global_prefetch_b8
        float x1 = gb[i*5+0], y1 = gb[i*5+1];
        float x2 = gb[i*5+2], y2 = gb[i*5+3], sc = gb[i*5+4];
        float ar = (x2 - x1) * (y2 - y1);
        ox1[i]=x1; oy1[i]=y1; ox2[i]=x2; oy2[i]=y2; osc[i]=sc; oar[i]=ar;
        skey[i] = (i < n) ? -ar : __builtin_inff();
        sidx[i] = i;
    }
    __syncthreads();

    // ---- Phase B: bitonic sort ascending by skey, carrying sidx ----
    for (int k = 2; k <= MBOX; k <<= 1) {
        for (int j = k >> 1; j > 0; j >>= 1) {
            for (int t = tid; t < MBOX/2; t += NTHREADS) {
                int i = ((t & ~(j - 1)) << 1) | (t & (j - 1));
                int p = i | j;
                bool up = ((i & k) == 0);
                float ki = skey[i], kp = skey[p];
                bool sw = up ? (ki > kp) : (ki < kp);
                if (sw) {
                    skey[i] = kp; skey[p] = ki;
                    int ti = sidx[i]; sidx[i] = sidx[p]; sidx[p] = ti;
                }
            }
            __syncthreads();
        }
    }

    // ---- Phase C: gather into sorted arrays ----
    for (int i = tid; i < MBOX; i += NTHREADS) {
        int o = sidx[i];
        sx1[i]=ox1[o]; sy1[i]=oy1[o]; sx2[i]=ox2[o]; sy2[i]=oy2[o]; sa[i]=oar[o];
        svv[i] = (o < n) ? 1 : 0;
    }
    __syncthreads();

    // ---- Phase D: containment sums via WMMA f32 <- f16 16x16x32 ----
    // A[i][k] = containment flag (0/1), B[k][n] = sa[k] broadcast over n.
    // 4 row-tiles batched per pass: each 32-column chunk is loaded from LDS
    // once (b128 loads) and reused by 4 WMMAs.
    {
        const int lane = tid & 31;
        const int wave = tid >> 5;     // 0..7
        const int h    = lane >> 4;    // lane half
        const int m    = lane & 15;    // A row within tile / D column N

        const float4* vx1 = (const float4*)sx1;
        const float4* vy1 = (const float4*)sy1;
        const float4* vx2 = (const float4*)sx2;
        const float4* vy2 = (const float4*)sy2;
        const float4* vsa = (const float4*)sa;
        const int4*   vvv = (const int4*)svv;

        for (int pass = 0; pass < 2; ++pass) {
            int   i0[4], ri[4];
            float rx1[4], ry1[4], rx2[4], ry2[4];
            v8f   acc[4];
            #pragma unroll
            for (int u = 0; u < 4; ++u) {
                int tile = wave + (pass * 4 + u) * (NTHREADS / 32);
                i0[u] = tile * 16;
                ri[u] = i0[u] + m;
                rx1[u] = sx1[ri[u]]; ry1[u] = sy1[ri[u]];
                rx2[u] = sx2[ri[u]]; ry2[u] = sy2[ri[u]];
                acc[u] = (v8f){};
            }

            for (int kb = 0; kb < MBOX; kb += 32) {
                // Column chunk -> registers (A-layout order: runs of 8)
                float cx1[16], cy1[16], cx2[16], cy2[16];
                int   cvl[16];
                const int q0 = (kb >> 2) + 2 * h;       // k = kb + 8h + [0..7]
                const int q1 = q0 + 4;                  // k = kb + 16 + 8h + [0..7]
                #pragma unroll
                for (int r2 = 0; r2 < 2; ++r2) {
                    int qq = (r2 == 0) ? q0 : q1;
                    int eo = r2 * 8;
                    float4 t0;
                    t0 = vx1[qq];   cx1[eo+0]=t0.x; cx1[eo+1]=t0.y; cx1[eo+2]=t0.z; cx1[eo+3]=t0.w;
                    t0 = vx1[qq+1]; cx1[eo+4]=t0.x; cx1[eo+5]=t0.y; cx1[eo+6]=t0.z; cx1[eo+7]=t0.w;
                    t0 = vy1[qq];   cy1[eo+0]=t0.x; cy1[eo+1]=t0.y; cy1[eo+2]=t0.z; cy1[eo+3]=t0.w;
                    t0 = vy1[qq+1]; cy1[eo+4]=t0.x; cy1[eo+5]=t0.y; cy1[eo+6]=t0.z; cy1[eo+7]=t0.w;
                    t0 = vx2[qq];   cx2[eo+0]=t0.x; cx2[eo+1]=t0.y; cx2[eo+2]=t0.z; cx2[eo+3]=t0.w;
                    t0 = vx2[qq+1]; cx2[eo+4]=t0.x; cx2[eo+5]=t0.y; cx2[eo+6]=t0.z; cx2[eo+7]=t0.w;
                    t0 = vy2[qq];   cy2[eo+0]=t0.x; cy2[eo+1]=t0.y; cy2[eo+2]=t0.z; cy2[eo+3]=t0.w;
                    t0 = vy2[qq+1]; cy2[eo+4]=t0.x; cy2[eo+5]=t0.y; cy2[eo+6]=t0.z; cy2[eo+7]=t0.w;
                    int4 ti;
                    ti = vvv[qq];   cvl[eo+0]=ti.x; cvl[eo+1]=ti.y; cvl[eo+2]=ti.z; cvl[eo+3]=ti.w;
                    ti = vvv[qq+1]; cvl[eo+4]=ti.x; cvl[eo+5]=ti.y; cvl[eo+6]=ti.z; cvl[eo+7]=ti.w;
                }

                // B fragment: b[e] = sa[kb + 8h + e] (e<8), sa[kb + 16 + 8h + e-8] (e>=8)
                v16h b;
                {
                    const int qb = (kb >> 2) + 2 * h;
                    float4 b0 = vsa[qb], b1 = vsa[qb+1], b2 = vsa[qb+4], b3 = vsa[qb+5];
                    b[0]=(_Float16)b0.x;  b[1]=(_Float16)b0.y;  b[2]=(_Float16)b0.z;  b[3]=(_Float16)b0.w;
                    b[4]=(_Float16)b1.x;  b[5]=(_Float16)b1.y;  b[6]=(_Float16)b1.z;  b[7]=(_Float16)b1.w;
                    b[8]=(_Float16)b2.x;  b[9]=(_Float16)b2.y;  b[10]=(_Float16)b2.z; b[11]=(_Float16)b2.w;
                    b[12]=(_Float16)b3.x; b[13]=(_Float16)b3.y; b[14]=(_Float16)b3.z; b[15]=(_Float16)b3.w;
                }

                #pragma unroll
                for (int u = 0; u < 4; ++u) {
                    v16h a;
                    #pragma unroll
                    for (int e = 0; e < 16; ++e) {
                        int ka = kb + ((e < 8) ? e : e + 8) + 8 * h;  // A 16-bit 16x32 layout
                        bool f = (cx1[e] >= rx1[u]) & (cy1[e] >= ry1[u]) &
                                 (cx2[e] <= rx2[u]) & (cy2[e] <= ry2[u]) &
                                 (cvl[e] != 0)      & (ka != ri[u]);
                        a[e] = f ? (_Float16)1.0f : (_Float16)0.0f;
                    }
                    acc[u] = __builtin_amdgcn_wmma_f32_16x16x32_f16(
                                 false, a, false, b, (short)0, acc[u], false, false);
                }
            }

            // D layout: column N==0 lives in lanes 0 (M=r) and 16 (M=r+8)
            if (m == 0) {
                #pragma unroll
                for (int u = 0; u < 4; ++u) {
                    #pragma unroll
                    for (int r = 0; r < 8; ++r) {
                        int i = i0[u] + r + 8 * h;
                        float s = acc[u][r];
                        keepS[i] = (int)((s <= THRESH * (sa[i] + 1e-9f)) & (svv[i] != 0));
                    }
                }
            }
        }
    }
    __syncthreads();

    // ---- Phase E: scatter keep to original order ----
    for (int i = tid; i < MBOX; i += NTHREADS) keepO[sidx[i]] = keepS[i];
    __syncthreads();

    // Block prefix scan over keepO (4 elements per thread)
    const int b4 = tid * 4;
    int k0 = keepO[b4], k1 = keepO[b4+1], k2 = keepO[b4+2], k3 = keepO[b4+3];
    int my = k0 + k1 + k2 + k3;
    scan[tid] = my;
    __syncthreads();
    for (int off = 1; off < NTHREADS; off <<= 1) {
        int v = (tid >= off) ? scan[tid - off] : 0;
        __syncthreads();
        scan[tid] += v;
        __syncthreads();
    }
    const int excl  = scan[tid] - my;
    const int total = scan[NTHREADS - 1];

    float* gout = out_boxes + (size_t)g * MBOX * 5;
    if (total > 0) {
        int p = excl;
        int kk[4] = {k0, k1, k2, k3};
        #pragma unroll
        for (int u = 0; u < 4; ++u) {
            int t4 = b4 + u;
            if (kk[u]) {
                gout[p*5+0] = ox1[t4]; gout[p*5+1] = oy1[t4];
                gout[p*5+2] = ox2[t4]; gout[p*5+3] = oy2[t4];
                gout[p*5+4] = osc[t4];
                ++p;
            }
        }
        for (int e = total * 5 + tid; e < MBOX * 5; e += NTHREADS) gout[e] = 0.0f;
    } else {
        for (int e = tid; e < MBOX * 5; e += NTHREADS) gout[e] = gb[e];
    }
    if (tid == 0) out_cnt[g] = total;
}

extern "C" void kernel_launch(void* const* d_in, const int* in_sizes, int n_in,
                              void* d_out, int out_size, void* d_ws, size_t ws_size,
                              hipStream_t stream) {
    const float* boxes  = (const float*)d_in[0];     // [T,C,M,5] f32
    const int*   nboxes = (const int*)d_in[1];       // [T,C] i32
    const int    groups = in_sizes[1];               // T*C = 128
    float* outb = (float*)d_out;
    int*   outc = (int*)((float*)d_out + (size_t)groups * MBOX * 5);
    BoxPromptFilter_kernel<<<groups, NTHREADS, 0, stream>>>(boxes, nboxes, outb, outc);
}
